// ValueNetork_67997922230665
// MI455X (gfx1250) — compile-verified, matrix operand order checked by reading, not run on hardware
//
#include <hip/hip_runtime.h>
#include <hip/hip_bf16.h>

// ---- problem dims ----
#define BB   8
#define LL   1024
#define DMD  256
#define DSD  16
#define DCD  4
#define DID  512      // d_inner
#define DTRD 16
#define ROWS (BB * LL)   // 8192
#define XDL  64          // padded x_dbl leading dim (48 -> 64)

typedef __attribute__((ext_vector_type(16))) __bf16 v16bf;
typedef __attribute__((ext_vector_type(8)))  float  v8f;

union Frag { uint4 u[2]; v16bf v; };

// ---------------- elementwise converts ----------------
__global__ void cvt_f32_bf16(const float* __restrict__ in,
                             __hip_bfloat16* __restrict__ out, int n) {
  int i = blockIdx.x * blockDim.x + threadIdx.x;
  if (i < n) out[i] = __float2bfloat16(in[i]);
}

// out is (Rout x Cout); in is (Rin x Cin); zero-pad outside [Rin,Cin)
__global__ void cvt_pad_bf16(const float* __restrict__ in,
                             __hip_bfloat16* __restrict__ out,
                             int Rin, int Cin, int Rout, int Cout) {
  int i = blockIdx.x * blockDim.x + threadIdx.x;
  if (i >= Rout * Cout) return;
  int r = i / Cout, c = i % Cout;
  float v = (r < Rin && c < Cin) ? in[(size_t)r * Cin + c] : 0.0f;
  out[i] = __float2bfloat16(v);
}

// out[c*R + r] = in[r*C + c]  (store W^T so B-columns become contiguous rows)
__global__ void transpose_cvt(const float* __restrict__ in,
                              __hip_bfloat16* __restrict__ out, int R, int C) {
  int i = blockIdx.x * blockDim.x + threadIdx.x;
  if (i >= R * C) return;
  int r = i / C, c = i % C;
  out[(size_t)c * R + r] = __float2bfloat16(in[(size_t)r * C + c]);
}

// ---------------- bf16 WMMA GEMM:  C[M,N] = A[M,K] * Bc[N,K]^T ----------------
// A row-major (lda halves/row). Bc holds column n of B as contiguous row n.
// Requirements: M%32==0, N%32==0, K%32==0, lda/ldb multiples of 8 halves.
// Each wave computes a 32x32 tile (2x2 WMMA tiles) -> 2x fragment reuse, 4-deep ILP.
__global__ __launch_bounds__(256) void gemm_bf16_wmma(
    const __hip_bfloat16* __restrict__ A, int lda,
    const __hip_bfloat16* __restrict__ Bc, int ldb,
    float* __restrict__ Cf32, __hip_bfloat16* __restrict__ Cbf16,
    const float* __restrict__ bias, int act,
    int M, int N, int K) {
  const int lane = threadIdx.x & 31;
  const int wave = threadIdx.x >> 5;
  const int tN = N >> 5;                      // 32-wide tiles
  const int tM = M >> 5;
  const int t = blockIdx.x * 8 + wave;        // wave-uniform tile id
  if (t >= tM * tN) return;                   // whole-wave exit; EXEC=all-1 for WMMA
  const int tm = (t / tN) << 5, tn = (t % tN) << 5;
  const int hi  = lane >> 4;                  // 0: K {0..7,16..23}, 1: K {8..15,24..31}
  const int l16 = lane & 15;
  const int g0  = hi ? 8 : 0;
  const __hip_bfloat16* A0 = A  + (size_t)(tm + l16) * lda + g0;
  const __hip_bfloat16* A1 = A0 + (size_t)16 * lda;
  const __hip_bfloat16* B0 = Bc + (size_t)(tn + l16) * ldb + g0;
  const __hip_bfloat16* B1 = B0 + (size_t)16 * ldb;

  v8f c00 = {}, c01 = {}, c10 = {}, c11 = {};
  for (int kb = 0; kb < K; kb += 32) {
    Frag a0, a1, b0, b1;
    a0.u[0] = *(const uint4*)(A0 + kb);
    a0.u[1] = *(const uint4*)(A0 + kb + 16);
    a1.u[0] = *(const uint4*)(A1 + kb);
    a1.u[1] = *(const uint4*)(A1 + kb + 16);
    b0.u[0] = *(const uint4*)(B0 + kb);
    b0.u[1] = *(const uint4*)(B0 + kb + 16);
    b1.u[0] = *(const uint4*)(B1 + kb);
    b1.u[1] = *(const uint4*)(B1 + kb + 16);
    c00 = __builtin_amdgcn_wmma_f32_16x16x32_bf16(false, a0.v, false, b0.v, (short)0, c00, false, false);
    c01 = __builtin_amdgcn_wmma_f32_16x16x32_bf16(false, a0.v, false, b1.v, (short)0, c01, false, false);
    c10 = __builtin_amdgcn_wmma_f32_16x16x32_bf16(false, a1.v, false, b0.v, (short)0, c10, false, false);
    c11 = __builtin_amdgcn_wmma_f32_16x16x32_bf16(false, a1.v, false, b1.v, (short)0, c11, false, false);
  }

  auto store_tile = [&](const v8f& acc, int mb, int nb) {
    const int n  = nb + l16;
    const float bv = bias ? bias[n] : 0.0f;
    #pragma unroll
    for (int r = 0; r < 8; ++r) {
      int m = mb + r + (hi ? 8 : 0);
      float v = acc[r] + bv;
      if (act == 1) v = fmaxf(v, 0.0f);
      size_t idx = (size_t)m * N + n;
      if (Cf32)  Cf32[idx]  = v;
      if (Cbf16) Cbf16[idx] = __float2bfloat16(v);
    }
  };
  store_tile(c00, tm,      tn);
  store_tile(c01, tm,      tn + 16);
  store_tile(c10, tm + 16, tn);
  store_tile(c11, tm + 16, tn + 16);
}

// ---------------- causal depthwise conv (DC=4) + SiLU ----------------
__global__ void conv_silu(const float* __restrict__ xz,      // (ROWS, 2*DID), x = cols [0,DID)
                          const float* __restrict__ conv_w,  // (DID, DCD)
                          const float* __restrict__ conv_b,
                          float* __restrict__ xs_f32,
                          __hip_bfloat16* __restrict__ xs_bf16) {
  int idx = blockIdx.x * blockDim.x + threadIdx.x;   // over ROWS*DID
  if (idx >= ROWS * DID) return;
  int d  = idx % DID;
  int bl = idx / DID;
  int l  = bl % LL;
  float acc = conv_b[d];
  #pragma unroll
  for (int k = 0; k < DCD; ++k) {
    int ls = l - (DCD - 1) + k;
    if (ls >= 0)
      acc += xz[(size_t)(bl - (l - ls)) * (2 * DID) + d] * conv_w[d * DCD + k];
  }
  float s = acc / (1.0f + __expf(-acc));             // silu
  xs_f32[idx]  = s;
  xs_bf16[idx] = __float2bfloat16(s);
}

// ---------------- selective scan: 16 lanes per (b,d) channel ----------------
__global__ __launch_bounds__(256) void scan_kernel(
    const float* __restrict__ dtpre,   // (ROWS, DID), dt_proj bias already added
    const float* __restrict__ x_dbl,   // (ROWS, XDL): [0,16)=dtr, [16,32)=B, [32,48)=C
    const float* __restrict__ xs,      // (ROWS, DID) silu(conv(x))
    const float* __restrict__ xz,      // (ROWS, 2*DID), z = cols [DID, 2*DID)
    const float* __restrict__ A_log,   // (DID, DSD)
    const float* __restrict__ Dv,      // (DID,)
    __hip_bfloat16* __restrict__ ybf)  // (ROWS, DID) gated output
{
  const int lane = threadIdx.x & 31;
  const int wgid = blockIdx.x * (blockDim.x >> 5) + (threadIdx.x >> 5);
  const int c = wgid * 2 + (lane >> 4);              // channel = b*DID + d
  if (c >= BB * DID) return;
  const int b = c / DID, d = c % DID, n = lane & 15;

  const float Aval = -__expf(A_log[d * DSD + n]);
  const float Dd   = Dv[d];
  const float* dtp = dtpre + (size_t)b * LL * DID + d;
  const float* xsp = xs    + (size_t)b * LL * DID + d;
  const float* zp  = xz    + (size_t)b * LL * (2 * DID) + DID + d;
  const float* xd  = x_dbl + (size_t)b * LL * XDL;

  float h = 0.0f;
  for (int l = 0; l < LL; ++l) {
    float dtv = dtp[(size_t)l * DID];
    float xv  = xsp[(size_t)l * DID];
    float Bn  = xd[l * XDL + 16 + n];
    float Cn  = xd[l * XDL + 32 + n];
    float dt  = (dtv > 20.0f) ? dtv : log1pf(__expf(dtv));   // softplus
    float dA  = __expf(dt * Aval);
    h = fmaf(dA, h, dt * Bn * xv);
    float r = h * Cn;                                        // reduce over 16 lanes
    r += __shfl_xor(r, 8);
    r += __shfl_xor(r, 4);
    r += __shfl_xor(r, 2);
    r += __shfl_xor(r, 1);
    if (n == 0) {
      float zv = zp[(size_t)l * (2 * DID)];
      float y  = r + xv * Dd;
      float g  = zv / (1.0f + __expf(-zv));                  // silu(z)
      ybf[(size_t)(b * LL + l) * DID + d] = __float2bfloat16(y * g);
    }
  }
}

// ---------------- final head: out = h2 @ w3 + b3 (N=1) ----------------
__global__ void head_kernel(const __hip_bfloat16* __restrict__ h2,
                            const float* __restrict__ w3,
                            const float* __restrict__ b3,
                            float* __restrict__ out, int rows) {
  int r = blockIdx.x * blockDim.x + threadIdx.x;
  if (r >= rows) return;
  const __hip_bfloat16* p = h2 + (size_t)r * 128;
  float acc = b3[0];
  #pragma unroll 4
  for (int j = 0; j < 128; ++j) acc += __bfloat162float(p[j]) * w3[j];
  out[r] = acc;
}

// ---------------- host launch ----------------
static inline int cdiv(int a, int b) { return (a + b - 1) / b; }

extern "C" void kernel_launch(void* const* d_in, const int* in_sizes, int n_in,
                              void* d_out, int out_size, void* d_ws, size_t ws_size,
                              hipStream_t stream) {
  const float* state     = (const float*)d_in[0];
  const float* in_proj_w = (const float*)d_in[1];
  const float* conv_w    = (const float*)d_in[2];
  const float* conv_b    = (const float*)d_in[3];
  const float* x_proj_w  = (const float*)d_in[4];
  const float* dt_proj_w = (const float*)d_in[5];
  const float* dt_proj_b = (const float*)d_in[6];
  const float* A_log     = (const float*)d_in[7];
  const float* Dvec      = (const float*)d_in[8];
  const float* out_proj_w= (const float*)d_in[9];
  const float* w1 = (const float*)d_in[10]; const float* b1 = (const float*)d_in[11];
  const float* w2 = (const float*)d_in[12]; const float* b2 = (const float*)d_in[13];
  const float* w3 = (const float*)d_in[14]; const float* b3 = (const float*)d_in[15];
  float* outp = (float*)d_out;

  char* ws = (char*)d_ws;
  auto carve = [&](size_t bytes) -> char* {
    char* p = ws; ws += (bytes + 255) & ~(size_t)255; return p;
  };
  typedef __hip_bfloat16 bf;
  bf* state_bf  = (bf*)carve((size_t)ROWS * DMD * 2);
  bf* inproj_bf = (bf*)carve((size_t)(2 * DID) * DMD * 2);
  bf* xproj_bf  = (bf*)carve((size_t)XDL * DID * 2);          // padded 48->64 rows
  bf* dtproj_bf = (bf*)carve((size_t)DID * 32 * 2);           // padded K 16->32
  bf* oproj_bf  = (bf*)carve((size_t)DMD * DID * 2);
  bf* w1T_bf    = (bf*)carve((size_t)256 * DMD * 2);
  bf* w2T_bf    = (bf*)carve((size_t)128 * 256 * 2);
  float* xz     = (float*)carve((size_t)ROWS * 2 * DID * 4);
  float* xs_f32 = (float*)carve((size_t)ROWS * DID * 4);
  bf*    xs_bf  = (bf*)carve((size_t)ROWS * DID * 2);
  float* xdbl_f = (float*)carve((size_t)ROWS * XDL * 4);
  bf*    xdbl_bf= (bf*)carve((size_t)ROWS * XDL * 2);
  float* dtpre  = (float*)carve((size_t)ROWS * DID * 4);
  bf*    y_bf   = (bf*)carve((size_t)ROWS * DID * 2);
  bf*    out_bf = (bf*)carve((size_t)ROWS * DMD * 2);
  bf*    h1_bf  = (bf*)carve((size_t)ROWS * 256 * 2);
  bf*    h2_bf  = (bf*)carve((size_t)ROWS * 128 * 2);
  (void)ws_size; (void)in_sizes; (void)n_in; (void)out_size;

  // 1) weight / activation converts (+ zero padding so every GEMM has K%32==0, N%32==0)
  { int n = ROWS * DMD;     cvt_f32_bf16<<<cdiv(n,256),256,0,stream>>>(state,      state_bf,  n); }
  { int n = 2 * DID * DMD;  cvt_f32_bf16<<<cdiv(n,256),256,0,stream>>>(in_proj_w,  inproj_bf, n); }
  { int n = DMD * DID;      cvt_f32_bf16<<<cdiv(n,256),256,0,stream>>>(out_proj_w, oproj_bf,  n); }
  { int n = XDL * DID;      cvt_pad_bf16<<<cdiv(n,256),256,0,stream>>>(x_proj_w,  xproj_bf,  48, DID, XDL, DID); }
  { int n = DID * 32;       cvt_pad_bf16<<<cdiv(n,256),256,0,stream>>>(dt_proj_w, dtproj_bf, DID, DTRD, DID, 32); }
  transpose_cvt<<<cdiv(DMD*256,256),256,0,stream>>>(w1, w1T_bf, DMD, 256);
  transpose_cvt<<<cdiv(256*128,256),256,0,stream>>>(w2, w2T_bf, 256, 128);

  // 2) in_proj: xz[8192,1024] = state @ in_proj_w^T
  { int tiles = (ROWS/32) * ((2*DID)/32);
    gemm_bf16_wmma<<<cdiv(tiles,8),256,0,stream>>>(state_bf, DMD, inproj_bf, DMD,
        xz, nullptr, nullptr, 0, ROWS, 2*DID, DMD); }

  // 3) causal depthwise conv + SiLU
  { int n = ROWS * DID;
    conv_silu<<<cdiv(n,256),256,0,stream>>>(xz, conv_w, conv_b, xs_f32, xs_bf); }

  // 4) x_proj: x_dbl[8192,64] = xs @ x_proj_w_pad^T   (cols 48..63 are zeros)
  { int tiles = (ROWS/32) * (XDL/32);
    gemm_bf16_wmma<<<cdiv(tiles,8),256,0,stream>>>(xs_bf, DID, xproj_bf, DID,
        xdbl_f, xdbl_bf, nullptr, 0, ROWS, XDL, DID); }

  // 5) dt_proj: dtpre[8192,512] = x_dbl[:, :32] @ dtproj_pad^T + dt_proj_b
  //    (dtproj cols 16..31 are zero, so the B-part of x_dbl contributes nothing)
  { int tiles = (ROWS/32) * (DID/32);
    gemm_bf16_wmma<<<cdiv(tiles,8),256,0,stream>>>(xdbl_bf, XDL, dtproj_bf, 32,
        dtpre, nullptr, dt_proj_b, 0, ROWS, DID, 32); }

  // 6) selective scan (softplus, state recurrence, C-contraction, D skip, z-gate)
  scan_kernel<<<(BB*DID)/2/8, 256, 0, stream>>>(dtpre, xdbl_f, xs_f32, xz,
                                                A_log, Dvec, y_bf);

  // 7) out_proj: out[8192,256] = y @ out_proj_w^T
  { int tiles = (ROWS/32) * (DMD/32);
    gemm_bf16_wmma<<<cdiv(tiles,8),256,0,stream>>>(y_bf, DID, oproj_bf, DID,
        nullptr, out_bf, nullptr, 0, ROWS, DMD, DID); }

  // 8) MLP head
  { int tiles = (ROWS/32) * (256/32);
    gemm_bf16_wmma<<<cdiv(tiles,8),256,0,stream>>>(out_bf, DMD, w1T_bf, DMD,
        nullptr, h1_bf, b1, 1, ROWS, 256, DMD); }
  { int tiles = (ROWS/32) * (128/32);
    gemm_bf16_wmma<<<cdiv(tiles,8),256,0,stream>>>(h1_bf, 256, w2T_bf, 256,
        nullptr, h2_bf, b2, 1, ROWS, 128, 256); }
  head_kernel<<<cdiv(ROWS,256),256,0,stream>>>(h2_bf, w3, b3, outp, ROWS);
}
